// UserItem_Aggregator_22419729285143
// MI455X (gfx1250) — compile-verified, matrix operand order checked by reading, not run on hardware
//
#include <hip/hip_runtime.h>
#include <hip/hip_bf16.h>

#define BB 8192
#define LLEN 50
#define DD 64

typedef _Float16 h16;
typedef __attribute__((ext_vector_type(16))) _Float16 v16h;
typedef __attribute__((ext_vector_type(8)))  _Float16 v8h;
typedef __attribute__((ext_vector_type(8)))  float    v8f;

// LDS row strides (halves): multiples of 8 (16B-aligned rows for b128),
// stride/2 mod 64 gives gap-4 bank bases across the 16 fragment lanes.
#define S128 136
#define S64  72

static __device__ __forceinline__ v16h v16h_zero() {
    v16h z;
#pragma unroll
    for (int i = 0; i < 16; ++i) z[i] = (h16)0.0f;
    return z;
}

static __device__ __forceinline__ v16h combine8(v8h a, v8h b) {
    return __builtin_shufflevector(a, b, 0,1,2,3,4,5,6,7,8,9,10,11,12,13,14,15);
}

// A-fragment for v_wmma_f32_16x16x32_f16 built straight from a global fp32 row.
// Lane layout: halves 0..7 -> K = hi*8 + 0..7 ; halves 8..15 -> K = 16 + hi*8 + 0..7.
static __device__ __forceinline__ v16h afrag_f32(const float* p, int hi) {
    const float* q0 = p + hi * 8;
    const float* q1 = p + 16 + hi * 8;
    float4 a = *(const float4*)(q0);
    float4 b = *(const float4*)(q0 + 4);
    float4 c = *(const float4*)(q1);
    float4 d = *(const float4*)(q1 + 4);
    v16h r;
    r[0]  = (h16)a.x; r[1]  = (h16)a.y; r[2]  = (h16)a.z; r[3]  = (h16)a.w;
    r[4]  = (h16)b.x; r[5]  = (h16)b.y; r[6]  = (h16)b.z; r[7]  = (h16)b.w;
    r[8]  = (h16)c.x; r[9]  = (h16)c.y; r[10] = (h16)c.z; r[11] = (h16)c.w;
    r[12] = (h16)d.x; r[13] = (h16)d.y; r[14] = (h16)d.z; r[15] = (h16)d.w;
    return r;
}

// A- or B-fragment from an f16 LDS matrix stored with contiguous K per row.
static __device__ __forceinline__ v16h ldsfrag(const h16* base, int row, int stride,
                                               int k0, int hi) {
    const h16* p = base + row * stride + k0 + hi * 8;
    v8h lo = *(const v8h*)(p);
    v8h hh = *(const v8h*)(p + 16);
    return combine8(lo, hh);
}

// ---- prep: fp32 weights -> f16, transposed (row = out-col n, contiguous k) in ws
__global__ void prep_weights(const float* __restrict__ W1, const float* __restrict__ W2,
                             const float* __restrict__ A1, const float* __restrict__ A2,
                             h16* __restrict__ ws) {
    int i = blockIdx.x * 256 + threadIdx.x;
    if (i < 8192) {                       // W1t [64][128]
        int n = i >> 7, k = i & 127;
        ws[n * 128 + k] = (h16)W1[k * 64 + n];
    } else if (i < 12288) {               // W2t [64][64]
        int j = i - 8192; int n = j >> 6, k = j & 63;
        ws[8192 + n * 64 + k] = (h16)W2[k * 64 + n];
    } else if (i < 16384) {               // A1 top half, transposed [64][64]
        int j = i - 12288; int n = j >> 6, k = j & 63;
        ws[12288 + n * 64 + k] = (h16)A1[k * 64 + n];
    } else if (i < 20480) {               // A2t [64][64]
        int j = i - 16384; int n = j >> 6, k = j & 63;
        ws[16384 + n * 64 + k] = (h16)A2[k * 64 + n];
    }
}

// ---- main: one block per batch row
__global__ __launch_bounds__(128)
void user_item_agg(const int* __restrict__ nodes, const int* __restrict__ history,
                   const int* __restrict__ rating, const int* __restrict__ maskp,
                   const float* __restrict__ item_emb, const float* __restrict__ user_emb,
                   const float* __restrict__ opinion_emb,
                   const float* __restrict__ b1, const float* __restrict__ b2,
                   const float* __restrict__ A1, const float* __restrict__ a1,
                   const float* __restrict__ a2b, const float* __restrict__ A3,
                   const float* __restrict__ a3,
                   const h16* __restrict__ ws, float* __restrict__ out) {
    __shared__ h16 W1t[64 * S128];     // GEMM1 B (transposed W1)
    __shared__ h16 W2t[64 * S64];
    __shared__ h16 A1tt[64 * S64];
    __shared__ h16 A2t[64 * S64];
    __shared__ h16 buf1[64 * S64];     // h, later a
    __shared__ h16 Ih[64 * S64];       // inter (kept to the end)
    __shared__ float ufs[64], uvecs[64], b1s[64], b2s[64], a2bs[64], A3s[64];
    __shared__ float scoresS[64], attS[64];
    __shared__ int hidx_s[64], ridx_s[64], msk_s[64];
    __shared__ float a3s;

    const int b    = blockIdx.x;
    const int tid  = threadIdx.x;
    const int w    = tid >> 5;
    const int lane = tid & 31;
    const int ln   = lane & 15;
    const int hi   = lane >> 4;

    // --- phase 0: weights ws->LDS (b128 copies), indices, scalars
    const h16* wsW1 = ws;
    const h16* wsW2 = ws + 8192;
    const h16* wsA1 = ws + 12288;
    const h16* wsA2 = ws + 16384;
    for (int i = tid; i < 64 * 16; i += 128) {
        int n = i >> 4, ck = i & 15;
        *(v8h*)&W1t[n * S128 + ck * 8] = *(const v8h*)&wsW1[n * 128 + ck * 8];
    }
    for (int i = tid; i < 64 * 8; i += 128) {
        int n = i >> 3, ck = i & 7;
        *(v8h*)&W2t[n * S64 + ck * 8]  = *(const v8h*)&wsW2[n * 64 + ck * 8];
        *(v8h*)&A1tt[n * S64 + ck * 8] = *(const v8h*)&wsA1[n * 64 + ck * 8];
        *(v8h*)&A2t[n * S64 + ck * 8]  = *(const v8h*)&wsA2[n * 64 + ck * 8];
    }
    if (tid < 64) {
        if (tid < LLEN) {
            hidx_s[tid] = history[b * LLEN + tid];
            ridx_s[tid] = rating[b * LLEN + tid];
            msk_s[tid]  = maskp[b * LLEN + tid];
        } else {
            hidx_s[tid] = 0; ridx_s[tid] = 0; msk_s[tid] = 0;
        }
        ufs[tid]  = user_emb[(size_t)nodes[b] * DD + tid];
        b1s[tid]  = b1[tid];
        b2s[tid]  = b2[tid];
        a2bs[tid] = a2b[tid];
        A3s[tid]  = A3[tid];
    }
    if (tid == 0) a3s = a3[0];
    __syncthreads();

    // --- phase 1: rank-1 part of attention input: uvec = u_rep @ A1[64:128] + a1
    if (tid < 64) {
        float acc = a1[tid];
#pragma unroll 8
        for (int d2 = 0; d2 < 64; ++d2) acc += ufs[d2] * A1[(64 + d2) * 64 + tid];
        uvecs[tid] = acc;
    }
    __syncthreads();

    // --- GEMM1: h = relu([hist,rate] @ W1 + b1), A-frags straight from gathers
    const int mrow = w * 16 + ln;
    v16h af[4];
    if (mrow < LLEN) {
        const float* hp = item_emb    + (size_t)hidx_s[mrow] * DD;
        const float* rp = opinion_emb + (size_t)ridx_s[mrow] * DD;
        af[0] = afrag_f32(hp, hi);
        af[1] = afrag_f32(hp + 32, hi);
        af[2] = afrag_f32(rp, hi);
        af[3] = afrag_f32(rp + 32, hi);
    } else {
        af[0] = af[1] = af[2] = af[3] = v16h_zero();
    }
#pragma unroll
    for (int nt = 0; nt < 4; ++nt) {
        v8f c;
        float bv = b1s[nt * 16 + ln];
#pragma unroll
        for (int r = 0; r < 8; ++r) c[r] = bv;
#pragma unroll
        for (int kt = 0; kt < 4; ++kt) {
            v16h bf = ldsfrag(W1t, nt * 16 + ln, S128, kt * 32, hi);
            c = __builtin_amdgcn_wmma_f32_16x16x32_f16(false, af[kt], false, bf,
                                                       (short)0, c, false, false);
        }
#pragma unroll
        for (int r = 0; r < 8; ++r) {
            int m = w * 16 + r + 8 * hi;
            buf1[m * S64 + nt * 16 + ln] = (h16)fmaxf(c[r], 0.0f);
        }
    }
    __syncthreads();

    // --- GEMM2: inter = relu(h @ W2 + b2)
    {
        v16h a0 = ldsfrag(buf1, mrow, S64, 0, hi);
        v16h a1f = ldsfrag(buf1, mrow, S64, 32, hi);
#pragma unroll
        for (int nt = 0; nt < 4; ++nt) {
            v8f c;
            float bv = b2s[nt * 16 + ln];
#pragma unroll
            for (int r = 0; r < 8; ++r) c[r] = bv;
            v16h bf0 = ldsfrag(W2t, nt * 16 + ln, S64, 0, hi);
            c = __builtin_amdgcn_wmma_f32_16x16x32_f16(false, a0, false, bf0,
                                                       (short)0, c, false, false);
            v16h bf1 = ldsfrag(W2t, nt * 16 + ln, S64, 32, hi);
            c = __builtin_amdgcn_wmma_f32_16x16x32_f16(false, a1f, false, bf1,
                                                       (short)0, c, false, false);
#pragma unroll
            for (int r = 0; r < 8; ++r) {
                int m = w * 16 + r + 8 * hi;
                Ih[m * S64 + nt * 16 + ln] = (h16)fmaxf(c[r], 0.0f);
            }
        }
    }
    __syncthreads();

    // --- GEMM3: a = relu(inter @ A1_top + uvec)   (uvec = u@A1_bot + a1)
    {
        v16h a0 = ldsfrag(Ih, mrow, S64, 0, hi);
        v16h a1f = ldsfrag(Ih, mrow, S64, 32, hi);
#pragma unroll
        for (int nt = 0; nt < 4; ++nt) {
            v8f c;
            float bv = uvecs[nt * 16 + ln];
#pragma unroll
            for (int r = 0; r < 8; ++r) c[r] = bv;
            v16h bf0 = ldsfrag(A1tt, nt * 16 + ln, S64, 0, hi);
            c = __builtin_amdgcn_wmma_f32_16x16x32_f16(false, a0, false, bf0,
                                                       (short)0, c, false, false);
            v16h bf1 = ldsfrag(A1tt, nt * 16 + ln, S64, 32, hi);
            c = __builtin_amdgcn_wmma_f32_16x16x32_f16(false, a1f, false, bf1,
                                                       (short)0, c, false, false);
#pragma unroll
            for (int r = 0; r < 8; ++r) {
                int m = w * 16 + r + 8 * hi;
                buf1[m * S64 + nt * 16 + ln] = (h16)fmaxf(c[r], 0.0f);
            }
        }
    }
    __syncthreads();

    // --- GEMM4: a2 = relu(a @ A2 + a2b); fold scores = a2 @ A3 into epilogue
    {
        v16h a0 = ldsfrag(buf1, mrow, S64, 0, hi);
        v16h a1f = ldsfrag(buf1, mrow, S64, 32, hi);
        float srow[8];
#pragma unroll
        for (int r = 0; r < 8; ++r) srow[r] = 0.0f;
#pragma unroll
        for (int nt = 0; nt < 4; ++nt) {
            v8f c;
            float bv = a2bs[nt * 16 + ln];
#pragma unroll
            for (int r = 0; r < 8; ++r) c[r] = bv;
            v16h bf0 = ldsfrag(A2t, nt * 16 + ln, S64, 0, hi);
            c = __builtin_amdgcn_wmma_f32_16x16x32_f16(false, a0, false, bf0,
                                                       (short)0, c, false, false);
            v16h bf1 = ldsfrag(A2t, nt * 16 + ln, S64, 32, hi);
            c = __builtin_amdgcn_wmma_f32_16x16x32_f16(false, a1f, false, bf1,
                                                       (short)0, c, false, false);
            float a3w = A3s[nt * 16 + ln];
#pragma unroll
            for (int r = 0; r < 8; ++r) srow[r] += fmaxf(c[r], 0.0f) * a3w;
        }
#pragma unroll
        for (int r = 0; r < 8; ++r) {
            float s = srow[r];
            s += __shfl_xor(s, 1, 32);
            s += __shfl_xor(s, 2, 32);
            s += __shfl_xor(s, 4, 32);
            s += __shfl_xor(s, 8, 32);
            if (ln == 0) {
                int m = w * 16 + r + 8 * hi;
                if (m < LLEN) scoresS[m] = (msk_s[m] > 0) ? (s + a3s) : -1e9f;
            }
        }
    }
    __syncthreads();

    // --- masked softmax over L=50
    if (tid < 64) attS[tid] = (tid < LLEN) ? scoresS[tid] : -3.0e38f;
    __syncthreads();
    for (int s2 = 32; s2 > 0; s2 >>= 1) {
        if (tid < s2) attS[tid] = fmaxf(attS[tid], attS[tid + s2]);
        __syncthreads();
    }
    float mx = attS[0];
    __syncthreads();
    if (tid < 64) {
        float e = (tid < LLEN) ? __expf(scoresS[tid] - mx) : 0.0f;
        scoresS[tid] = e;
        attS[tid] = e;
    }
    __syncthreads();
    for (int s2 = 32; s2 > 0; s2 >>= 1) {
        if (tid < s2) attS[tid] += attS[tid + s2];
        __syncthreads();
    }
    float inv = 1.0f / attS[0];
    __syncthreads();
    if (tid < 64) scoresS[tid] *= inv;
    __syncthreads();

    // --- weighted aggregation: out[b,d] = sum_l att[l] * inter[l,d]
    if (tid < 64) {
        float acc = 0.0f;
#pragma unroll 10
        for (int l = 0; l < LLEN; ++l) acc += scoresS[l] * (float)Ih[l * S64 + tid];
        out[(size_t)b * DD + tid] = acc;
    }
}

extern "C" void kernel_launch(void* const* d_in, const int* in_sizes, int n_in,
                              void* d_out, int out_size, void* d_ws, size_t ws_size,
                              hipStream_t stream) {
    const int*   nodes       = (const int*)d_in[0];
    const int*   history     = (const int*)d_in[1];
    const int*   rating      = (const int*)d_in[2];
    const int*   maskp       = (const int*)d_in[3];
    const float* item_emb    = (const float*)d_in[4];
    const float* user_emb    = (const float*)d_in[5];
    const float* opinion_emb = (const float*)d_in[6];
    const float* W1          = (const float*)d_in[7];
    const float* b1          = (const float*)d_in[8];
    const float* W2          = (const float*)d_in[9];
    const float* b2          = (const float*)d_in[10];
    const float* A1          = (const float*)d_in[11];
    const float* a1          = (const float*)d_in[12];
    const float* A2          = (const float*)d_in[13];
    const float* a2b         = (const float*)d_in[14];
    const float* A3          = (const float*)d_in[15];
    const float* a3          = (const float*)d_in[16];
    float*       out         = (float*)d_out;
    h16*         ws          = (h16*)d_ws;

    prep_weights<<<80, 256, 0, stream>>>(W1, W2, A1, A2, ws);
    user_item_agg<<<BB, 128, 0, stream>>>(nodes, history, rating, maskp,
                                          item_emb, user_emb, opinion_emb,
                                          b1, b2, A1, a1, a2b, A3, a3, ws, out);
}
